// OIMLoss_12429635354663
// MI455X (gfx1250) — matrix operand order repaired and checked by reading, not compiled, not run on hardware
//
#include <hip/hip_runtime.h>
#include <math.h>

#define NFEAT      256
#define NPIDS      5532
#define CQSZ       5000
#define NCOLS      (NPIDS + CQSZ)          /* 10532 */
#define IGNORE_IDX 5554
#define OIM_SCALAR 30.0f
#define NWAVES     8
#define ROWS_PER_WG (16 * NWAVES)          /* 128 */
#define NTILES     ((NCOLS + 15) / 16)     /* 659 */
#define COL_BYTES  (NFEAT * 2)             /* 512 B per column (bf16) */
#define COL_PITCH  (COL_BYTES + 16)        /* 528 B: bank-conflict pad */
#define TILE_LDS   (16 * COL_PITCH)        /* 8448 B per buffer */

typedef __attribute__((ext_vector_type(16))) __bf16 v16bf;
typedef __attribute__((ext_vector_type(8)))  float  v8f;
typedef __attribute__((ext_vector_type(4)))  int    v4i;

typedef __attribute__((address_space(1))) v4i* as1_v4i_p;
typedef __attribute__((address_space(3))) v4i* as3_v4i_p;

union FragBF { v16bf v; unsigned int d[8]; };

// ---------------- CDNA5 async global->LDS path (guarded) ----------------
#if defined(__has_builtin)
#  if __has_builtin(__builtin_amdgcn_global_load_async_to_lds_b128)
#    define HAVE_ASYNC_LDS 1
#  endif
#endif
#ifndef HAVE_ASYNC_LDS
#  define HAVE_ASYNC_LDS 0
#endif

__device__ __forceinline__ void async_copy16(const void* g, void* l) {
#if HAVE_ASYNC_LDS
  __builtin_amdgcn_global_load_async_to_lds_b128(
      (as1_v4i_p)(void*)g, (as3_v4i_p)l, /*offset=*/0, /*cpol=*/0);
#else
  *(uint4*)l = *(const uint4*)g;
#endif
}

template <int N>
__device__ __forceinline__ void wait_async() {
#if HAVE_ASYNC_LDS
#  if __has_builtin(__builtin_amdgcn_s_wait_asynccnt)
  __builtin_amdgcn_s_wait_asynccnt(N);
#  else
  asm volatile("s_wait_asynccnt %0" :: "i"(N) : "memory");
#  endif
#endif
}

// ---------------- fp32 -> packed bf16 (native cvt, RNE) ----------------
__device__ __forceinline__ unsigned int pk2bf(float a, float b) {
  unsigned short lo = __builtin_bit_cast(unsigned short, (__bf16)a);
  unsigned short hi = __builtin_bit_cast(unsigned short, (__bf16)b);
  return (unsigned int)lo | ((unsigned int)hi << 16);
}

// One-shot conversion of W=[lut;cq] and X to bf16 in workspace.
__global__ void oim_convert(const float* __restrict__ X,
                            const float* __restrict__ lut,
                            const float* __restrict__ cq,
                            unsigned int* __restrict__ Wb,
                            unsigned int* __restrict__ Xb, int nrows)
{
  const int wpairs = NCOLS * NFEAT / 2;
  const int xpairs = nrows * NFEAT / 2;
  const int total  = wpairs + xpairs;
  for (int i = blockIdx.x * blockDim.x + threadIdx.x; i < total;
       i += gridDim.x * blockDim.x) {
    if (i < wpairs) {
      const int e = i * 2;
      const float* src = (e < NPIDS * NFEAT) ? (lut + e)
                                             : (cq + (e - NPIDS * NFEAT));
      Wb[i] = pk2bf(src[0], src[1]);
    } else {
      const int e = (i - wpairs) * 2;
      Xb[i - wpairs] = pk2bf(X[e], X[e + 1]);
    }
  }
}

// Fused bf16-WMMA GEMM + fixed-max softmax-denominator + NLL partials.
// 128 rows per workgroup; B tiles double-buffered in LDS via async copy.
__global__ __launch_bounds__(256) void oim_main(
    const float* __restrict__ X, const int* __restrict__ lbl,
    const float* __restrict__ lut, const float* __restrict__ cq,
    const unsigned short* __restrict__ Wb,
    const unsigned short* __restrict__ Xb,
    float* __restrict__ part)
{
  __shared__ unsigned char lds_b[2][TILE_LDS];
  __shared__ float lds_S[NWAVES][16];
  __shared__ float s_nll[ROWS_PER_WG];
  __shared__ float s_val[ROWS_PER_WG];

  const int tid  = threadIdx.x;
  const int wave = tid >> 5;
  const int lane = tid & 31;
  const int half = lane >> 4;
  const int l16  = lane & 15;
  const int rowbase = blockIdx.x * ROWS_PER_WG + wave * 16;  // this wave's 16 rows

  // ---- A fragments (16 rows x 256 K, bf16) register-resident for the kernel.
  // ISA 16-bit A 16x32 layout: lane(l16)=row M; dword pairs at strided K.
  FragBF afrag[8];
  {
    const unsigned short* xrow = Xb + (size_t)(rowbase + l16) * NFEAT;
    #pragma unroll
    for (int kc = 0; kc < 8; ++kc) {
      #pragma unroll
      for (int p = 0; p < 8; ++p) {
        const int e = 2 * p;
        const int klocal = (e & 7) + (half ? 8 : 0) + ((e >> 3) ? 16 : 0);
        afrag[kc].d[p] = *(const unsigned int*)(xrow + kc * 32 + klocal);
      }
    }
  }

  // Cooperative prefetch of one 16-column B tile (8 KB) into LDS buffer.
  auto prefetch = [&](int jt, int buf) {
    #pragma unroll
    for (int c = 0; c < 2; ++c) {
      const int idx  = tid + c * 256;      // 512 x 16B chunks
      const int col  = idx >> 5;
      const int koff = (idx & 31) * 16;
      int gcol = jt * 16 + col;
      gcol = (gcol < NCOLS) ? gcol : (NCOLS - 1);   // clamp last tile
      async_copy16((const unsigned char*)Wb + (size_t)gcol * COL_BYTES + koff,
                   &lds_b[buf][col * COL_PITCH + koff]);
    }
  };

  // ---- per-lane softmax-denominator state (fixed max = 30: |dot| <= 1).
  float rS[8];
  #pragma unroll
  for (int r = 0; r < 8; ++r) rS[r] = 0.0f;

  prefetch(0, 0);
  for (int jt = 0; jt < NTILES; ++jt) {
    const int buf = jt & 1;
    if (jt + 1 < NTILES) { prefetch(jt + 1, buf ^ 1); wait_async<2>(); }
    else                 { wait_async<0>(); }
    __syncthreads();                       // tile jt visible to all waves

    const unsigned char* bcol = &lds_b[buf][l16 * COL_PITCH];
    v8f acc = {};
    #pragma unroll
    for (int kc = 0; kc < 8; ++kc) {
      // B KxN layout: lane = column N; K 8h..8h+7 then 16+8h..16+8h+7 per lane.
      FragBF b;
      const uint4* p0 = (const uint4*)(bcol + kc * 64 + half * 16);
      uint4 q0 = p0[0], q1 = p0[2];
      b.d[0] = q0.x; b.d[1] = q0.y; b.d[2] = q0.z; b.d[3] = q0.w;
      b.d[4] = q1.x; b.d[5] = q1.y; b.d[6] = q1.z; b.d[7] = q1.w;
      acc = __builtin_amdgcn_wmma_f32_16x16x32_bf16(
          false, afrag[kc].v, false, b.v, (short)0, acc, false, false);
    }

    if (jt * 16 + l16 < NCOLS) {
      #pragma unroll
      for (int r = 0; r < 8; ++r)
        rS[r] += __expf(fmaf(acc[r], OIM_SCALAR, -OIM_SCALAR)); // exp(30*dot-30)
    }
    __syncthreads();                       // done reading buf before reuse
  }

  // ---- sum across the 16 column-lanes of each half-wave.
  #pragma unroll
  for (int r = 0; r < 8; ++r) {
    float s = rS[r];
    #pragma unroll
    for (int msk = 1; msk < 16; msk <<= 1) s += __shfl_xor(s, msk, 32);
    if (l16 == 0) lds_S[wave][r + half * 8] = s;   // rows 0-7 / 8-15 of wave
  }
  __syncthreads();

  // ---- one thread per row: lse + exact fp32 label logit.
  if (tid < ROWS_PER_WG) {
    const int row = blockIdx.x * ROWS_PER_WG + tid;
    const float S   = lds_S[tid >> 4][tid & 15];
    const float lse = __logf(S) + OIM_SCALAR;
    const int lab = lbl[row];
    float nll = 0.0f, val = 0.0f;
    if (lab != IGNORE_IDX) {
      const float* wr = (lab < NPIDS) ? (lut + (size_t)lab * NFEAT)
                                      : (cq + (size_t)(lab - NPIDS) * NFEAT);
      const float* ar = X + (size_t)row * NFEAT;
      float dot = 0.0f;
      #pragma unroll 8
      for (int k = 0; k < NFEAT; ++k) dot = fmaf(ar[k], wr[k], dot);
      nll = lse - OIM_SCALAR * dot;
      val = 1.0f;
    }
    s_nll[tid] = nll;
    s_val[tid] = val;
  }
  __syncthreads();

  // ---- deterministic per-block partial (fixed order).
  if (tid == 0) {
    float a = 0.0f, c = 0.0f;
    for (int i = 0; i < ROWS_PER_WG; ++i) { a += s_nll[i]; c += s_val[i]; }
    part[2 * blockIdx.x + 0] = a;
    part[2 * blockIdx.x + 1] = c;
  }
}

__global__ void oim_finalize(const float* __restrict__ part,
                             float* __restrict__ out, int nblocks)
{
  if (blockIdx.x == 0 && threadIdx.x == 0) {
    float s = 0.0f, c = 0.0f;
    for (int i = 0; i < nblocks; ++i) { s += part[2 * i]; c += part[2 * i + 1]; }
    out[0] = s / fmaxf(c, 1.0f);
  }
}

extern "C" void kernel_launch(void* const* d_in, const int* in_sizes, int n_in,
                              void* d_out, int out_size, void* d_ws, size_t ws_size,
                              hipStream_t stream) {
  const float* X   = (const float*)d_in[0];
  const int*   lbl = (const int*)d_in[1];
  const float* lut = (const float*)d_in[2];
  const float* cq  = (const float*)d_in[3];
  float* out = (float*)d_out;

  const int nrows = in_sizes[1];           // 4096
  const int nblk  = nrows / ROWS_PER_WG;   // 32

  char* ws = (char*)d_ws;
  const size_t WBYTES = (size_t)NCOLS * NFEAT * 2;  // 5.39 MB bf16 W
  const size_t XBYTES = (size_t)nrows * NFEAT * 2;  // 2.10 MB bf16 X
  unsigned int* Wb32 = (unsigned int*)ws;
  unsigned int* Xb32 = (unsigned int*)(ws + WBYTES);
  float* part = (float*)(ws + WBYTES + XBYTES);

  oim_convert<<<1024, 256, 0, stream>>>(X, lut, cq, Wb32, Xb32, nrows);
  oim_main<<<nblk, 256, 0, stream>>>(X, lbl, lut, cq,
                                     (const unsigned short*)Wb32,
                                     (const unsigned short*)Xb32, part);
  oim_finalize<<<1, 32, 0, stream>>>(part, out, nblk);
}
